// ReflectiveCore_53772990546467
// MI455X (gfx1250) — matrix operand order repaired and checked by reading, not compiled, not run on hardware
//
#include <hip/hip_runtime.h>
#include <hip/hip_bf16.h>
#include <math.h>

#define D_    1024
#define NHEAD_ 16
#define DH_    64
#define NL_     2
#define DFF_ 4096
#define B_      4
#define T_    512
#define C_     64
#define L_    576
#define V_  16000
#define S_      4

typedef __bf16 bf16_t;
typedef __bf16 v16bf __attribute__((ext_vector_type(16)));
typedef __bf16 v8bf  __attribute__((ext_vector_type(8)));
typedef float  v8f   __attribute__((ext_vector_type(8)));
typedef float  v4f   __attribute__((ext_vector_type(4)));

static __device__ __forceinline__ bf16_t f2bf(float f) { return (bf16_t)f; }

// A-fragment from bf16 memory: two contiguous 8-elem (16B) chunks.
static __device__ __forceinline__ v16bf frag2x8(const bf16_t* __restrict__ p0,
                                                const bf16_t* __restrict__ p1) {
    v8bf a = *(const v8bf*)p0;
    v8bf b = *(const v8bf*)p1;
    return __builtin_shufflevector(a, b, 0, 1, 2, 3, 4, 5, 6, 7,
                                         8, 9, 10, 11, 12, 13, 14, 15);
}
// B-fragment from bf16 memory: one contiguous 16-elem (32B) chunk.
static __device__ __forceinline__ v16bf frag16(const bf16_t* __restrict__ p) {
    return *(const v16bf*)p;
}
// Fragment from fp32 memory (16 contiguous floats), cvt in-register.
static __device__ __forceinline__ v16bf frag16f(const float* __restrict__ p) {
    v4f x0 = *(const v4f*)(p);
    v4f x1 = *(const v4f*)(p + 4);
    v4f x2 = *(const v4f*)(p + 8);
    v4f x3 = *(const v4f*)(p + 12);
    v16bf f;
#pragma unroll
    for (int e = 0; e < 4; ++e) {
        f[e]      = f2bf(x0[e]);
        f[4 + e]  = f2bf(x1[e]);
        f[8 + e]  = f2bf(x2[e]);
        f[12 + e] = f2bf(x3[e]);
    }
    return f;
}

// ---------------------------------------------------------------------------
// bf16 GEMM: Y[M,N] = act(X[M,K] @ W[N,K]^T + bias[N]); X,W bf16; acc f32.
// Wave tile 32(M) x 64(N), 4 waves stacked in M -> 128x64 per block.
// Optional fp32 and/or bf16 output streams.
// ---------------------------------------------------------------------------
__global__ __launch_bounds__(128) void k_gemm_bf(
    const bf16_t* __restrict__ X, long lda,
    const bf16_t* __restrict__ W, int ldw,
    const float* __restrict__ bias,
    float* __restrict__ Yf, bf16_t* __restrict__ Ybf, long ldc,
    int M, int N, int K, int relu)
{
    const int lane = threadIdx.x;
    const int wave = threadIdx.y;
    const int g    = lane >> 4;
    const int ln16 = lane & 15;

    const int n0 = blockIdx.x * 64;
    const int m0 = blockIdx.y * 128 + wave * 32;

    v8f zero = {};
    v8f acc[2][4];
#pragma unroll
    for (int i = 0; i < 2; ++i)
#pragma unroll
        for (int j = 0; j < 4; ++j) acc[i][j] = zero;

    long arow[2];
#pragma unroll
    for (int i = 0; i < 2; ++i) {
        int r = m0 + i * 16 + ln16;
        if (r >= M) r = M - 1;          // clamp (safe load; store is guarded)
        arow[i] = (long)r * lda;
    }
    long brow[4];
#pragma unroll
    for (int j = 0; j < 4; ++j)
        brow[j] = (long)(n0 + j * 16 + ln16) * ldw;

    for (int k0 = 0; k0 < K; k0 += 32) {
        v16bf a[2], b[4];
#pragma unroll
        for (int i = 0; i < 2; ++i)
            a[i] = frag2x8(X + arow[i] + k0 + g * 16,
                           X + arow[i] + k0 + g * 16 + 8);
#pragma unroll
        for (int j = 0; j < 4; ++j)
            b[j] = frag16(W + brow[j] + k0 + g * 16);
#pragma unroll
        for (int i = 0; i < 2; ++i)
#pragma unroll
            for (int j = 0; j < 4; ++j)
                acc[i][j] = __builtin_amdgcn_wmma_f32_16x16x32_bf16(
                    false, a[i], false, b[j], (short)0, acc[i][j], false, false);
    }

#pragma unroll
    for (int i = 0; i < 2; ++i) {
#pragma unroll
        for (int j = 0; j < 4; ++j) {
            int col = n0 + j * 16 + ln16;
            float bs = bias ? bias[col] : 0.0f;
#pragma unroll
            for (int r = 0; r < 8; ++r) {
                int row = m0 + i * 16 + r + 8 * g;
                if (row < M) {
                    float v = acc[i][j][r] + bs;
                    if (relu) v = fmaxf(v, 0.0f);
                    if (Yf)  Yf[(long)row * ldc + col] = v;
                    if (Ybf) Ybf[(long)row * ldc + col] = f2bf(v);
                }
            }
        }
    }
}

// ---------------------------------------------------------------------------
// Logits GEMM: out[M<=16, N] = Xbf[M,K] @ Wf32[N,K]^T + bias. W streamed from
// HBM exactly once; 4 waves of a block cover 4 different 64-col strips.
// ---------------------------------------------------------------------------
__global__ __launch_bounds__(128) void k_gemm_logits(
    const bf16_t* __restrict__ X, long lda,
    const float* __restrict__ W, int ldw,
    const float* __restrict__ bias,
    float* __restrict__ Y, long ldc,
    int M, int N, int K)
{
    const int lane = threadIdx.x;
    const int wave = threadIdx.y;
    const int g    = lane >> 4;
    const int ln16 = lane & 15;

    const int n0 = blockIdx.x * 256 + wave * 64;

    v8f zero = {};
    v8f acc[4];
#pragma unroll
    for (int j = 0; j < 4; ++j) acc[j] = zero;

    int ar = ln16 < M ? ln16 : M - 1;
    const long arow = (long)ar * lda;
    long brow[4];
#pragma unroll
    for (int j = 0; j < 4; ++j)
        brow[j] = (long)(n0 + j * 16 + ln16) * ldw;

    for (int k0 = 0; k0 < K; k0 += 32) {
        if (k0 + 32 < K)
            __builtin_prefetch(W + brow[0] + k0 + 32, 0, 0);
        v16bf a = frag2x8(X + arow + k0 + g * 16,
                          X + arow + k0 + g * 16 + 8);
#pragma unroll
        for (int j = 0; j < 4; ++j) {
            v16bf b = frag16f(W + brow[j] + k0 + g * 16);
            acc[j] = __builtin_amdgcn_wmma_f32_16x16x32_bf16(
                false, a, false, b, (short)0, acc[j], false, false);
        }
    }

#pragma unroll
    for (int j = 0; j < 4; ++j) {
        int col = n0 + j * 16 + ln16;
        float bs = bias[col];
#pragma unroll
        for (int r = 0; r < 8; ++r) {
            int row = r + 8 * g;
            if (row < M)
                Y[(long)row * ldc + col] = acc[j][r] + bs;
        }
    }
}

// ---------------------------------------------------------------------------
// Flash attention over bf16 Q/K/V: one wave per (16-query tile, head, batch).
// ---------------------------------------------------------------------------
__global__ __launch_bounds__(32) void k_attn(
    const bf16_t* __restrict__ Q,  long qs, int qoff,
    const bf16_t* __restrict__ Kb, long ks, int koff,
    const bf16_t* __restrict__ Vb, long vs, int voff,
    bf16_t* __restrict__ O, long os,
    int Lq, int Lk, float scale)
{
    const int lane = threadIdx.x;
    const int g    = lane >> 4;
    const int ln16 = lane & 15;
    const int q0 = blockIdx.x * 16;
    const int h  = blockIdx.y;
    const int b  = blockIdx.z;
    const int hb = h * DH_;

    __shared__ bf16_t sP[16 * 32];

    long qrow = (long)(b * Lq + q0 + ln16) * qs + qoff + hb;
    v16bf aq[2];
#pragma unroll
    for (int s = 0; s < 2; ++s)
        aq[s] = frag2x8(Q + qrow + s * 32 + g * 16,
                        Q + qrow + s * 32 + g * 16 + 8);

    v8f zero = {};
    v8f o[4];
    float m_run[8], l_run[8];
#pragma unroll
    for (int j = 0; j < 4; ++j) o[j] = zero;
#pragma unroll
    for (int r = 0; r < 8; ++r) { m_run[r] = -1e30f; l_run[r] = 0.0f; }

    for (int kt = 0; kt < Lk; kt += 32) {
        // scores: 16 queries x 32 keys
        v8f sc[2];
#pragma unroll
        for (int half = 0; half < 2; ++half) {
            long krow = (long)(b * Lk + kt + half * 16 + ln16) * ks + koff + hb;
            v8f c = zero;
#pragma unroll
            for (int s = 0; s < 2; ++s) {
                v16bf bk = frag16(Kb + krow + s * 32 + g * 16);
                c = __builtin_amdgcn_wmma_f32_16x16x32_bf16(
                        false, aq[s], false, bk, (short)0, c, false, false);
            }
#pragma unroll
            for (int r = 0; r < 8; ++r) c[r] *= scale;
            sc[half] = c;
        }

        // online softmax (row r+8g spans one 16-lane half)
        float alpha[8];
#pragma unroll
        for (int r = 0; r < 8; ++r) {
            float mx = fmaxf(sc[0][r], sc[1][r]);
            mx = fmaxf(mx, __shfl_xor(mx, 1, 32));
            mx = fmaxf(mx, __shfl_xor(mx, 2, 32));
            mx = fmaxf(mx, __shfl_xor(mx, 4, 32));
            mx = fmaxf(mx, __shfl_xor(mx, 8, 32));
            float mn = fmaxf(m_run[r], mx);
            alpha[r] = __expf(m_run[r] - mn);
            m_run[r] = mn;
            float p0 = __expf(sc[0][r] - mn);
            float p1 = __expf(sc[1][r] - mn);
            sc[0][r] = p0; sc[1][r] = p1;
            float sm = p0 + p1;
            sm += __shfl_xor(sm, 1, 32);
            sm += __shfl_xor(sm, 2, 32);
            sm += __shfl_xor(sm, 4, 32);
            sm += __shfl_xor(sm, 8, 32);
            l_run[r] = l_run[r] * alpha[r] + sm;
        }

        // P: C-layout -> LDS (16x32 bf16) -> A-layout fragment
        __syncthreads();
#pragma unroll
        for (int r = 0; r < 8; ++r) {
            int row = r + 8 * g;
            sP[row * 32 + ln16]      = f2bf(sc[0][r]);
            sP[row * 32 + 16 + ln16] = f2bf(sc[1][r]);
        }
        __syncthreads();
        v16bf ap;
#pragma unroll
        for (int e = 0; e < 8; ++e) {
            ap[e]     = sP[ln16 * 32 + g * 16 + e];
            ap[8 + e] = sP[ln16 * 32 + g * 16 + 8 + e];
        }

        // rescale accumulators, O += P @ V
#pragma unroll
        for (int j = 0; j < 4; ++j) {
#pragma unroll
            for (int r = 0; r < 8; ++r) o[j][r] *= alpha[r];
            v16bf bv;
#pragma unroll
            for (int e = 0; e < 16; ++e) {
                int key = kt + g * 16 + e;
                bv[e] = Vb[(long)(b * Lk + key) * vs + voff + hb + j * 16 + ln16];
            }
            o[j] = __builtin_amdgcn_wmma_f32_16x16x32_bf16(
                       false, ap, false, bv, (short)0, o[j], false, false);
        }
    }

#pragma unroll
    for (int j = 0; j < 4; ++j) {
#pragma unroll
        for (int r = 0; r < 8; ++r) {
            int row = q0 + r + 8 * g;
            O[(long)(b * Lq + row) * os + hb + j * 16 + ln16] =
                f2bf(o[j][r] / l_run[r]);
        }
    }
}

// ---------------------------------------------------------------------------
// Fused residual + LayerNorm: fp32 stream out (residual) + bf16 stream out
// (next GEMM operand).
// ---------------------------------------------------------------------------
__global__ __launch_bounds__(256) void k_ln(
    const float* __restrict__ X, const float* __restrict__ R,
    const float* __restrict__ s, const float* __restrict__ bta,
    float* __restrict__ Yf, bf16_t* __restrict__ Ybf)
{
    __shared__ float sbuf[256];
    const int tid = threadIdx.x;
    const long row = blockIdx.x;
    const float* x = X + row * D_;
    const float* r = R + row * D_;

    float v[4];
    float sum = 0.0f;
#pragma unroll
    for (int i = 0; i < 4; ++i) {
        int d = tid + i * 256;
        v[i] = x[d] + r[d];
        sum += v[i];
    }
    sbuf[tid] = sum; __syncthreads();
    for (int st = 128; st > 0; st >>= 1) {
        if (tid < st) sbuf[tid] += sbuf[tid + st];
        __syncthreads();
    }
    float mu = sbuf[0] * (1.0f / D_);
    __syncthreads();

    float vs = 0.0f;
#pragma unroll
    for (int i = 0; i < 4; ++i) { float d = v[i] - mu; vs += d * d; }
    sbuf[tid] = vs; __syncthreads();
    for (int st = 128; st > 0; st >>= 1) {
        if (tid < st) sbuf[tid] += sbuf[tid + st];
        __syncthreads();
    }
    float inv = rsqrtf(sbuf[0] * (1.0f / D_) + 1e-5f);

#pragma unroll
    for (int i = 0; i < 4; ++i) {
        int d = tid + i * 256;
        float y = (v[i] - mu) * inv * s[d] + bta[d];
        Yf[row * D_ + d] = y;
        Ybf[row * D_ + d] = f2bf(y);
    }
}

// ---------------------------------------------------------------------------
// Embedding gather: fp32 h (residual stream), bf16 h and bf16 mem (GEMM A).
// ---------------------------------------------------------------------------
__global__ __launch_bounds__(256) void k_embed(
    const int* __restrict__ token, const int* __restrict__ state,
    const float* __restrict__ prefix, const float* __restrict__ W_in,
    const float* __restrict__ b_in, const float* __restrict__ pos,
    float* __restrict__ h32, bf16_t* __restrict__ hbf, bf16_t* __restrict__ mem)
{
    const int l = blockIdx.x, b = blockIdx.y;
    const long base = (long)(b * L_ + l) * D_;
    if (l < C_) {
        for (int d = threadIdx.x; d < D_; d += 256) {
            float v = prefix[(long)(b * C_ + l) * D_ + d] + pos[l * D_ + d];
            h32[base + d] = v;
            bf16_t vb = f2bf(v);
            hbf[base + d] = vb; mem[base + d] = vb;
        }
    } else {
        const int idx = token[b * T_ + (l - C_)] * S_ + state[b];
        for (int d = threadIdx.x; d < D_; d += 256) {
            float v = W_in[(long)d * (V_ * S_) + idx] + b_in[d] + pos[l * D_ + d];
            h32[base + d] = v;
            bf16_t vb = f2bf(v);
            hbf[base + d] = vb; mem[base + d] = vb;
        }
    }
}

// fp32 -> bf16 bulk conversion (weights), 4 elems/thread.
__global__ __launch_bounds__(256) void k_cvt(
    const float* __restrict__ X, bf16_t* __restrict__ Y, long n)
{
    long i = ((long)blockIdx.x * 256 + threadIdx.x) * 4;
    if (i + 3 < n) {
        v4f x = *(const v4f*)(X + i);
        Y[i]     = f2bf(x[0]);
        Y[i + 1] = f2bf(x[1]);
        Y[i + 2] = f2bf(x[2]);
        Y[i + 3] = f2bf(x[3]);
    }
}

extern "C" void kernel_launch(void* const* d_in, const int* in_sizes, int n_in,
                              void* d_out, int out_size, void* d_ws, size_t ws_size,
                              hipStream_t stream) {
    const int*   token   = (const int*)  d_in[0];
    const int*   state   = (const int*)  d_in[1];
    const float* prefix  = (const float*)d_in[2];
    const float* W_in    = (const float*)d_in[3];
    const float* b_in    = (const float*)d_in[4];
    const float* pos     = (const float*)d_in[5];
    const float* sa_in_w = (const float*)d_in[6];
    const float* sa_in_b = (const float*)d_in[7];
    const float* sa_out_w= (const float*)d_in[8];
    const float* sa_out_b= (const float*)d_in[9];
    const float* ca_in_w = (const float*)d_in[10];
    const float* ca_in_b = (const float*)d_in[11];
    const float* ca_out_w= (const float*)d_in[12];
    const float* ca_out_b= (const float*)d_in[13];
    const float* ff1_w   = (const float*)d_in[14];
    const float* ff1_b   = (const float*)d_in[15];
    const float* ff2_w   = (const float*)d_in[16];
    const float* ff2_b   = (const float*)d_in[17];
    const float* ln1_s   = (const float*)d_in[18];
    const float* ln1_b   = (const float*)d_in[19];
    const float* ln2_s   = (const float*)d_in[20];
    const float* ln2_b   = (const float*)d_in[21];
    const float* ln3_s   = (const float*)d_in[22];
    const float* ln3_b   = (const float*)d_in[23];
    const float* W_out   = (const float*)d_in[24];
    const float* b_out   = (const float*)d_in[25];
    float* out = (float*)d_out;

    const long MTOT = (long)B_ * L_;     // 2304 rows

    char* p = (char*)d_ws;
    float* h32 = (float*)p;  p += (long)MTOT * D_ * 4;
    float* tmp = (float*)p;  p += (long)MTOT * D_ * 4;
    bf16_t* hbf  = (bf16_t*)p; p += (long)MTOT * D_ * 2;
    bf16_t* mem  = (bf16_t*)p; p += (long)MTOT * D_ * 2;
    bf16_t* qkv  = (bf16_t*)p; p += (long)MTOT * 3 * D_ * 2;
    bf16_t* kv   = (bf16_t*)p; p += (long)MTOT * 2 * D_ * 2;
    bf16_t* attn = (bf16_t*)p; p += (long)MTOT * D_ * 2;
    bf16_t* ff   = (bf16_t*)p; p += (long)MTOT * DFF_ * 2;
    bf16_t* w_sa_in  = (bf16_t*)p; p += (long)NL_ * 3 * D_ * D_ * 2;
    bf16_t* w_sa_out = (bf16_t*)p; p += (long)NL_ * D_ * D_ * 2;
    bf16_t* w_ca_in  = (bf16_t*)p; p += (long)NL_ * 3 * D_ * D_ * 2;
    bf16_t* w_ca_out = (bf16_t*)p; p += (long)NL_ * D_ * D_ * 2;
    bf16_t* w_ff1    = (bf16_t*)p; p += (long)NL_ * DFF_ * D_ * 2;
    bf16_t* w_ff2    = (bf16_t*)p; p += (long)NL_ * D_ * DFF_ * 2;

    dim3 gblk(32, 4, 1);
    auto gemm = [&](const bf16_t* X, long lda, const bf16_t* W, int ldw,
                    const float* bias, float* Yf, bf16_t* Ybf, long ldc,
                    int M, int N, int K, int relu) {
        dim3 grid(N / 64, (M + 127) / 128, 1);
        k_gemm_bf<<<grid, gblk, 0, stream>>>(X, lda, W, ldw, bias, Yf, Ybf, ldc,
                                             M, N, K, relu);
    };
    auto cvt = [&](const float* X, bf16_t* Y, long n) {
        k_cvt<<<(int)(n / 1024), 256, 0, stream>>>(X, Y, n);
    };

    // one-time weight conversion (64 MB bf16; L2-resident afterwards)
    cvt(sa_in_w,  w_sa_in,  (long)NL_ * 3 * D_ * D_);
    cvt(sa_out_w, w_sa_out, (long)NL_ * D_ * D_);
    cvt(ca_in_w,  w_ca_in,  (long)NL_ * 3 * D_ * D_);
    cvt(ca_out_w, w_ca_out, (long)NL_ * D_ * D_);
    cvt(ff1_w,    w_ff1,    (long)NL_ * DFF_ * D_);
    cvt(ff2_w,    w_ff2,    (long)NL_ * D_ * DFF_);

    k_embed<<<dim3(L_, B_), 256, 0, stream>>>(token, state, prefix, W_in, b_in,
                                              pos, h32, hbf, mem);

    const dim3 agrid(L_ / 16, NHEAD_, B_);
    for (int i = 0; i < NL_; ++i) {
        // ---- self attention ----
        gemm(hbf, D_, w_sa_in + (long)i * 3 * D_ * D_, D_, sa_in_b + i * 3 * D_,
             nullptr, qkv, 3 * D_, (int)MTOT, 3 * D_, D_, 0);
        k_attn<<<agrid, 32, 0, stream>>>(qkv, 3 * D_, 0,
                                         qkv, 3 * D_, D_,
                                         qkv, 3 * D_, 2 * D_,
                                         attn, D_, L_, L_, 0.125f);
        gemm(attn, D_, w_sa_out + (long)i * D_ * D_, D_, sa_out_b + i * D_,
             tmp, nullptr, D_, (int)MTOT, D_, D_, 0);
        k_ln<<<(int)MTOT, 256, 0, stream>>>(h32, tmp, ln1_s + i * D_,
                                            ln1_b + i * D_, h32, hbf);

        // ---- cross attention (K/V from mem) ----
        gemm(hbf, D_, w_ca_in + (long)i * 3 * D_ * D_, D_, ca_in_b + i * 3 * D_,
             nullptr, qkv, D_, (int)MTOT, D_, D_, 0);
        gemm(mem, D_, w_ca_in + (long)i * 3 * D_ * D_ + (long)D_ * D_, D_,
             ca_in_b + i * 3 * D_ + D_,
             nullptr, kv, 2 * D_, (int)MTOT, 2 * D_, D_, 0);
        k_attn<<<agrid, 32, 0, stream>>>(qkv, D_, 0,
                                         kv, 2 * D_, 0,
                                         kv, 2 * D_, D_,
                                         attn, D_, L_, L_, 0.125f);
        gemm(attn, D_, w_ca_out + (long)i * D_ * D_, D_, ca_out_b + i * D_,
             tmp, nullptr, D_, (int)MTOT, D_, D_, 0);
        k_ln<<<(int)MTOT, 256, 0, stream>>>(h32, tmp, ln2_s + i * D_,
                                            ln2_b + i * D_, h32, hbf);

        // ---- FFN ----
        gemm(hbf, D_, w_ff1 + (long)i * DFF_ * D_, D_, ff1_b + i * DFF_,
             nullptr, ff, DFF_, (int)MTOT, DFF_, D_, 1);
        gemm(ff, DFF_, w_ff2 + (long)i * D_ * DFF_, DFF_, ff2_b + i * D_,
             tmp, nullptr, D_, (int)MTOT, D_, DFF_, 0);
        k_ln<<<(int)MTOT, 256, 0, stream>>>(h32, tmp, ln3_s + i * D_,
                                            ln3_b + i * D_, h32, hbf);
    }

    // ---- final logits: hbf[b, L-1, :] @ W_out^T + b_out -> [B, V*S] ----
    k_gemm_logits<<<dim3((V_ * S_) / 256, 1, 1), gblk, 0, stream>>>(
        hbf + (long)(L_ - 1) * D_, (long)L_ * D_, W_out, D_, b_out,
        out, V_ * S_, B_, V_ * S_, D_);
}
